// FCN_MODEL_87763361727284
// MI455X (gfx1250) — matrix operand order repaired and checked by reading, not compiled, not run on hardware
//
#include <hip/hip_runtime.h>
#include <cmath>

typedef __attribute__((ext_vector_type(16))) _Float16 v16h;
typedef __attribute__((ext_vector_type(8)))  _Float16 v8h;
typedef __attribute__((ext_vector_type(8)))  float    v8f;

#define K_DIM     512
#define HID_DIM   256
#define TILE_M    16
#define WS_STRIDE 520   // 512 + 8 halves pad  -> bank spread for B reads
#define AS_STRIDE 520   // 512 + 8 halves pad  -> bank spread for A reads
#define XB_STRIDE 272   // 256 + 16 floats pad -> conflict-free LN reads
#define SMEM_BYTES ((HID_DIM*WS_STRIDE + TILE_M*AS_STRIDE)*2 + TILE_M*XB_STRIDE*4)

__global__ void w1_cvt_kernel(const float* __restrict__ W1,
                              _Float16* __restrict__ W1h, int n) {
  int i = blockIdx.x * blockDim.x + threadIdx.x;
  if (i < n) W1h[i] = (_Float16)W1[i];
}

__global__ __launch_bounds__(256)
void edge_mlp_kernel(const float* __restrict__ h_all,
                     const int* __restrict__ src,
                     const int* __restrict__ dst,
                     const _Float16* __restrict__ W1h,
                     const float* __restrict__ b1,
                     const float* __restrict__ W3,
                     const float* __restrict__ b3,
                     const float* __restrict__ gamma2,
                     const float* __restrict__ beta2,
                     float* __restrict__ out,
                     int Nn, int E, int numTiles) {
  extern __shared__ char smem_raw[];
  _Float16* Ws = (_Float16*)smem_raw;                 // 256 x 520 f16 (W1, padded rows)
  _Float16* As = Ws + HID_DIM * WS_STRIDE;            // 16 x 520 f16 (A tile)
  float*    Xb = (float*)(As + TILE_M * AS_STRIDE);   // 16 x 272 f32 (GEMM result)

  const int t    = threadIdx.x;
  const int w    = t >> 5;        // wave id 0..7 -> owns HID cols [32w, 32w+32)
  const int lane = t & 31;
  const int s    = t & 15;        // lane-in-edge for gather / LN phases
  const int m    = t >> 4;        // edge-in-tile 0..15 for gather / LN phases

  // ---- per-thread LayerNorm / W3 params for columns n = s + 16*i ----
  float rb1[16], rg[16], rbt[16], rw3[16];
#pragma unroll
  for (int i = 0; i < 16; ++i) {
    int n = s + (i << 4);
    rb1[i] = b1[n]; rg[i] = gamma2[n]; rbt[i] = beta2[n]; rw3[i] = W3[n];
  }
  const float b3v = b3[0];

  // ---- stage W1 (f16) into LDS once, with padded row stride ----
  for (int i = t; i < (HID_DIM * K_DIM) / 8; i += 256) {
    int el  = i << 3;
    int row = el >> 9;            // /512
    int col = el & (K_DIM - 1);
    *(v8h*)(Ws + row * WS_STRIDE + col) = *(const v8h*)(W1h + el);
  }

  // ---- WMMA fragment addressing (ISA 7.12.2, wave32) ----
  const int mrow = lane & 15;                 // A row
  const int kbA  = (lane < 16) ? 0 : 16;      // A K-base per half-wave
  const int kbB  = (lane < 16) ? 0 : 16;      // B K-base per half-wave
  const int nCol = (w << 5) + (lane & 15);    // B/C column for tile 0
  const _Float16* arow  = As + mrow * AS_STRIDE;
  const _Float16* brow0 = Ws + nCol * WS_STRIDE;
  const _Float16* brow1 = Ws + (nCol + 16) * WS_STRIDE;

  for (int tile = blockIdx.x; tile < numTiles; tile += gridDim.x) {
    __syncthreads();  // protect As (WMMA reads) & Xb (LN reads) of prior tile

    // ---- gather + scale + tanh -> f16 A tile ----
    int e = tile * TILE_M + m;
    if (e >= E) e = E - 1;
    const int si = src[e];
    const int di = dst[e];
    for (int i = 0; i < 32; ++i) {
      int k = s + (i << 4);                    // 0..511
      int l = k >> 7;                          // layer
      int c = k & 127;                         // concat position
      int idx = (c < 64) ? si : di;
      float v = h_all[((size_t)l * Nn + idx) * 64 + (c & 63)];
      float imp = (float)(l + 1) * 0.1f;       // imp = (l+1)/10
      As[m * AS_STRIDE + k] = (_Float16)tanhf(v * imp);
    }
    __syncthreads();

    // ---- GEMM: (16 x 512) x (512 x 32-per-wave), f16 WMMA, f32 acc ----
    v8f acc0 = {};
    v8f acc1 = {};
#pragma unroll 4
    for (int kt = 0; kt < K_DIM / 32; ++kt) {
      const int kb = kt * 32;
      v8h alo = *(const v8h*)(arow + kb + kbA);
      v8h ahi = *(const v8h*)(arow + kb + kbA + 8);
      v16h a = __builtin_shufflevector(alo, ahi,
                 0,1,2,3,4,5,6,7,8,9,10,11,12,13,14,15);
      v8h b0l = *(const v8h*)(brow0 + kb + kbB);
      v8h b0h = *(const v8h*)(brow0 + kb + kbB + 8);
      v16h bf0 = __builtin_shufflevector(b0l, b0h,
                 0,1,2,3,4,5,6,7,8,9,10,11,12,13,14,15);
      v8h b1l = *(const v8h*)(brow1 + kb + kbB);
      v8h b1h = *(const v8h*)(brow1 + kb + kbB + 8);
      v16h bf1 = __builtin_shufflevector(b1l, b1h,
                 0,1,2,3,4,5,6,7,8,9,10,11,12,13,14,15);
      acc0 = __builtin_amdgcn_wmma_f32_16x16x32_f16(
          false, a, false, bf0, (short)0, acc0, false, false);
      acc1 = __builtin_amdgcn_wmma_f32_16x16x32_f16(
          false, a, false, bf1, (short)0, acc1, false, false);
    }

    // ---- stage x (pre-bias) to LDS f32 ----
    {
      const int col   = (w << 5) + (lane & 15);
      const int rbase = (lane < 16) ? 0 : 8;
#pragma unroll
      for (int r = 0; r < 8; ++r) {
        Xb[(rbase + r) * XB_STRIDE + col]      = acc0[r];
        Xb[(rbase + r) * XB_STRIDE + col + 16] = acc1[r];
      }
    }
    __syncthreads();

    // ---- +b1, LayerNorm, ReLU, dot(W3), +b3 (16 lanes per edge) ----
    float xv[16];
    float sum = 0.f, sq = 0.f;
#pragma unroll
    for (int i = 0; i < 16; ++i) {
      float v = Xb[m * XB_STRIDE + s + (i << 4)] + rb1[i];
      xv[i] = v;
      sum += v;
      sq  += v * v;
    }
#pragma unroll
    for (int msk = 8; msk >= 1; msk >>= 1) {
      sum += __shfl_xor(sum, msk, 32);
      sq  += __shfl_xor(sq,  msk, 32);
    }
    const float mu   = sum * (1.0f / 256.0f);
    const float var  = sq * (1.0f / 256.0f) - mu * mu;
    const float rstd = rsqrtf(var + 1e-5f);
    float acc = 0.f;
#pragma unroll
    for (int i = 0; i < 16; ++i) {
      float y = (xv[i] - mu) * rstd * rg[i] + rbt[i];
      y = fmaxf(y, 0.0f);
      acc = fmaf(y, rw3[i], acc);
    }
#pragma unroll
    for (int msk = 8; msk >= 1; msk >>= 1) acc += __shfl_xor(acc, msk, 32);

    int eo = tile * TILE_M + m;
    if (s == 0 && eo < E) out[eo] = acc + b3v;
  }
}

extern "C" void kernel_launch(void* const* d_in, const int* in_sizes, int n_in,
                              void* d_out, int out_size, void* d_ws, size_t ws_size,
                              hipStream_t stream) {
  const float* h_all  = (const float*)d_in[0];
  const int*   src    = (const int*)  d_in[1];
  const int*   dst    = (const int*)  d_in[2];
  const float* W1     = (const float*)d_in[3];
  const float* b1     = (const float*)d_in[4];
  const float* W3     = (const float*)d_in[5];
  const float* b3     = (const float*)d_in[6];
  const float* gamma2 = (const float*)d_in[7];
  const float* beta2  = (const float*)d_in[8];
  float* out = (float*)d_out;

  const int Nn = in_sizes[0] / (4 * 64);     // L*N*D / (L*D)
  const int E  = in_sizes[1];
  const int numTiles = (E + TILE_M - 1) / TILE_M;

  _Float16* W1h = (_Float16*)d_ws;           // 256 KB scratch for f16 W1
  const int w1n = HID_DIM * K_DIM;

  w1_cvt_kernel<<<(w1n + 255) / 256, 256, 0, stream>>>(W1, W1h, w1n);

  int blocks = numTiles < 1024 ? numTiles : 1024;  // persistent blocks, 1/WGP (LDS-bound)
  edge_mlp_kernel<<<blocks, 256, SMEM_BYTES, stream>>>(
      h_all, src, dst, W1h, b1, W3, b3, gamma2, beta2, out, Nn, E, numTiles);
}
